// GNNWrapper_52501680226462
// MI455X (gfx1250) — matrix operand order repaired and checked by reading, not compiled, not run on hardware
//
#include <hip/hip_runtime.h>
#include <hip/hip_bf16.h>
#include <stdint.h>

#define NN 40000
#define EE 640000
#define CC 128
#define LL 3
#define BB 64

typedef __attribute__((ext_vector_type(16))) __bf16 v16bf;
typedef __attribute__((ext_vector_type(8)))  float  v8f;

union FragB { uint32_t u[8]; v16bf v; };
union FragC { float    f[8]; v8f   v; };

static __device__ __forceinline__ uint32_t f2bf(float x) {
    union { float f; uint32_t u; } c; c.f = x;
    uint32_t u = c.u;
    return ((u + 0x7FFFu + ((u >> 16) & 1u)) >> 16) & 0xFFFFu;  // RNE
}
static __device__ __forceinline__ uint32_t pack2bf(float lo, float hi) {
    return f2bf(lo) | (f2bf(hi) << 16);
}

// ---------------------------------------------------------------------------
// Pack fp32 weight matrices [128x128] (row = K/in, col = N/out) into the
// per-lane WMMA B-fragment layout for v_wmma_f32_16x16x32_bf16:
//   index = (((m*8 + colTile)*4 + kc)*32 + lane)*8 + j
//   lane<16 : N = colTile*16 + lane,     K = kc*32 + 2j, 2j+1
//   lane>=16: N = colTile*16 + lane-16,  K = kc*32 + 16 + 2j, +1
// ---------------------------------------------------------------------------
__global__ __launch_bounds__(256) void prep_weights(
    const float* __restrict__ W, uint32_t* __restrict__ Wp, int nmat)
{
    int idx = blockIdx.x * blockDim.x + threadIdx.x;
    int total = nmat * 8192;
    if (idx >= total) return;
    int j    = idx & 7;
    int lane = (idx >> 3) & 31;
    int kc   = (idx >> 8) & 3;
    int ct   = (idx >> 10) & 7;
    int m    = idx >> 13;
    int n  = ct * 16 + (lane & 15);
    int k0 = kc * 32 + ((lane >> 4) ? 16 : 0) + 2 * j;
    const float* Wm = W + (size_t)m * CC * CC;
    Wp[idx] = pack2bf(Wm[(size_t)k0 * CC + n], Wm[(size_t)(k0 + 1) * CC + n]);
}

// ---------------------------------------------------------------------------
// GEMM: Out[rows x 128] = A[rows x 128] @ W[128 x 128] (+bias)(+res)(relu)
// 256 threads = 8 waves; block computes 16 rows x 128 cols; grid = rows/16
// (all row counts are exact multiples of 16 -> no bounds checks, straight-line
// epilogue). SELF_INIT additionally writes Out2 = Out*sn[row] + bias2[col]
// (fused GCN self-loop init, saves a full read+write pass over hw).
// ---------------------------------------------------------------------------
template<bool HAS_BIAS, bool RELU, bool HAS_RES, bool SELF_INIT>
__global__ __launch_bounds__(256) void gemm_bf16(
    const float* __restrict__ A, const uint32_t* __restrict__ Wp,
    const float* __restrict__ bias, const float* __restrict__ res,
    const float* __restrict__ sn, const float* __restrict__ bias2,
    float* __restrict__ Out, float* __restrict__ Out2)
{
    __shared__ uint32_t atile[16 * 65];
    const int t = threadIdx.x;
    const int rowBase = blockIdx.x * 16;

    // Cooperative fill: 1024 packed uints (16 rows x 64 bf16-pairs)
#pragma unroll
    for (int i = 0; i < 4; ++i) {
        int uidx = t * 4 + i;
        int row = uidx >> 6, col = uidx & 63;
        const float* p = A + (size_t)(rowBase + row) * CC + col * 2;
        atile[row * 65 + col] = pack2bf(p[0], p[1]);
    }
    __syncthreads();

    const int wave = t >> 5;     // column tile 0..7
    const int lane = t & 31;
    const int half = lane >> 4;
    const int l16  = lane & 15;

    FragC acc;
#pragma unroll
    for (int j = 0; j < 8; ++j) acc.f[j] = 0.f;

    const uint32_t* wbase = Wp + (size_t)wave * 4 * 256;   // 256 uints per (ct,kc)
#pragma unroll
    for (int kc = 0; kc < 4; ++kc) {
        FragB a, b;
        const uint32_t* wk = wbase + kc * 256 + lane * 8;
#pragma unroll
        for (int j = 0; j < 8; ++j) b.u[j] = wk[j];
        // A fragment: lane<16 K-pairs {0..7}+kc*16 ; lane>=16 {8..15}+kc*16
        const int kb = kc * 16 + half * 8;
#pragma unroll
        for (int j = 0; j < 8; ++j) {
            int jj = j;
            a.u[j] = atile[l16 * 65 + kb + jj];
        }
        acc.v = __builtin_amdgcn_wmma_f32_16x16x32_bf16(
            false, a.v, false, b.v, (short)0, acc.v, false, false);
    }

    const int col = wave * 16 + l16;
    float bv = 0.f, b2v = 0.f;
    if (HAS_BIAS)  bv  = bias[col];
    if (SELF_INIT) b2v = bias2[col];
    float snv[8];
    if (SELF_INIT) {
#pragma unroll
        for (int j = 0; j < 8; ++j) snv[j] = sn[rowBase + half * 8 + j];
    }
#pragma unroll
    for (int j = 0; j < 8; ++j) {
        int row = rowBase + half * 8 + j;   // lanes 0-15 -> M=j, lanes 16-31 -> M=8+j
        float v = acc.f[j] + bv;
        if (HAS_RES) v += res[(size_t)row * CC + col];
        if (RELU)    v = fmaxf(v, 0.f);
        Out[(size_t)row * CC + col] = v;
        if (SELF_INIT) Out2[(size_t)row * CC + col] = v * snv[j] + b2v;
    }
}

// ---------------------------------------------------------------------------
// Degree / normalization
// ---------------------------------------------------------------------------
__global__ void deg_init(float* __restrict__ deg, int n) {
    int i = blockIdx.x * blockDim.x + threadIdx.x;
    if (i < n) deg[i] = 1.0f;                       // self-loop
}
__global__ void deg_count(const int* __restrict__ dst, float* __restrict__ deg, int nE) {
    int e = blockIdx.x * blockDim.x + threadIdx.x;
    if (e < nE) atomicAdd(&deg[dst[e]], 1.0f);
}
__global__ void deg_finish(const float* __restrict__ deg,
                           float* __restrict__ isq, float* __restrict__ sn, int n) {
    int i = blockIdx.x * blockDim.x + threadIdx.x;
    if (i < n) { float r = rsqrtf(deg[i]); isq[i] = r; sn[i] = r * r; }
}

// agg[dst] += hw[src] * isq[src]*isq[dst]   (thread = edge x 4 channels)
__global__ __launch_bounds__(256) void edge_aggregate(
    const int* __restrict__ src, const int* __restrict__ dst,
    const float* __restrict__ hw, const float* __restrict__ isq,
    float* __restrict__ agg, int nE)
{
    long long idx = (long long)blockIdx.x * blockDim.x + threadIdx.x;
    int e = (int)(idx >> 5), grp = (int)(idx & 31);
    if (e >= nE) return;
    int s = src[e], d = dst[e];
    float w = isq[s] * isq[d];
    int c = grp * 4;
    const float4 v = *(const float4*)(hw + (size_t)s * CC + c);
    float* ap = agg + (size_t)d * CC + c;
    atomicAdd(ap + 0, v.x * w);
    atomicAdd(ap + 1, v.y * w);
    atomicAdd(ap + 2, v.z * w);
    atomicAdd(ap + 3, v.w * w);
}

// ---------------------------------------------------------------------------
// LayerNorm over 128 channels, one wave32 per row (4 channels/lane).
// RELU_RES: out = relu(LN(x)) + res   (post-conv path), else out = LN(x).
// ---------------------------------------------------------------------------
template<bool RELU_RES>
__global__ __launch_bounds__(256) void layernorm128(
    const float* __restrict__ X, const float* __restrict__ g,
    const float* __restrict__ b, const float* __restrict__ res,
    float* __restrict__ Out)
{
    int wave = threadIdx.x >> 5, lane = threadIdx.x & 31;
    int row = blockIdx.x * 8 + wave;
    const float* x = X + (size_t)row * CC;
    float4 v = *(const float4*)(x + lane * 4);
    float s  = v.x + v.y + v.z + v.w;
    float sq = v.x * v.x + v.y * v.y + v.z * v.z + v.w * v.w;
#pragma unroll
    for (int off = 16; off > 0; off >>= 1) {
        s  += __shfl_xor(s,  off, 32);
        sq += __shfl_xor(sq, off, 32);
    }
    float mean = s * (1.f / 128.f);
    float var  = sq * (1.f / 128.f) - mean * mean;
    float rs   = rsqrtf(var + 1e-5f);
    float4 gg = *(const float4*)(g + lane * 4);
    float4 bb = *(const float4*)(b + lane * 4);
    float4 o;
    o.x = (v.x - mean) * rs * gg.x + bb.x;
    o.y = (v.y - mean) * rs * gg.y + bb.y;
    o.z = (v.z - mean) * rs * gg.z + bb.z;
    o.w = (v.w - mean) * rs * gg.w + bb.w;
    if (RELU_RES) {
        float4 r = *(const float4*)(res + (size_t)row * CC + lane * 4);
        o.x = fmaxf(o.x, 0.f) + r.x;
        o.y = fmaxf(o.y, 0.f) + r.y;
        o.z = fmaxf(o.z, 0.f) + r.z;
        o.w = fmaxf(o.w, 0.f) + r.w;
    }
    *(float4*)(Out + (size_t)row * CC + lane * 4) = o;
}

// ---------------------------------------------------------------------------
// Global mean pool + classifier tail
// ---------------------------------------------------------------------------
__global__ void zero_f(float* __restrict__ p, int n) {
    int i = blockIdx.x * blockDim.x + threadIdx.x;
    if (i < n) p[i] = 0.f;
}
__global__ void pool_count(const int* __restrict__ batch, float* __restrict__ cnt, int n) {
    int i = blockIdx.x * blockDim.x + threadIdx.x;
    if (i < n) atomicAdd(&cnt[batch[i]], 1.0f);
}
__global__ __launch_bounds__(256) void pool_sum(
    const float* __restrict__ h, const int* __restrict__ batch,
    float* __restrict__ g, int n)
{
    int idx = blockIdx.x * blockDim.x + threadIdx.x;
    int node = idx >> 5, grp = idx & 31;
    if (node >= n) return;
    int bb = batch[node];
    int c = grp * 4;
    const float4 v = *(const float4*)(h + (size_t)node * CC + c);
    float* gp = g + (size_t)bb * CC + c;
    atomicAdd(gp + 0, v.x);
    atomicAdd(gp + 1, v.y);
    atomicAdd(gp + 2, v.z);
    atomicAdd(gp + 3, v.w);
}
__global__ void pool_div(float* __restrict__ g, const float* __restrict__ cnt, int n) {
    int i = blockIdx.x * blockDim.x + threadIdx.x;
    if (i < n) g[i] /= fmaxf(cnt[i >> 7], 1.0f);
}
// out[b] = dot(t[b,:], w2) + b2 ; one wave per row
__global__ __launch_bounds__(256) void cls_out_kernel(
    const float* __restrict__ tin, const float* __restrict__ w2,
    const float* __restrict__ b2, float* __restrict__ out)
{
    int wave = threadIdx.x >> 5, lane = threadIdx.x & 31;
    int row = blockIdx.x * 8 + wave;
    const float* x = tin + (size_t)row * CC;
    float s = 0.f;
#pragma unroll
    for (int k = 0; k < 4; ++k) s += x[lane * 4 + k] * w2[lane * 4 + k];
#pragma unroll
    for (int off = 16; off > 0; off >>= 1) s += __shfl_xor(s, off, 32);
    if (lane == 0) out[row] = s + b2[0];
}

// ---------------------------------------------------------------------------
extern "C" void kernel_launch(void* const* d_in, const int* in_sizes, int n_in,
                              void* d_out, int out_size, void* d_ws, size_t ws_size,
                              hipStream_t stream)
{
    const float* x        = (const float*)d_in[0];
    const int*   ei       = (const int*)  d_in[1];
    const int*   batch    = (const int*)  d_in[2];
    const float* pre_w    = (const float*)d_in[3];
    const float* pre_b    = (const float*)d_in[4];
    const float* conv_w   = (const float*)d_in[5];
    const float* conv_b   = (const float*)d_in[6];
    const float* ln_g     = (const float*)d_in[7];
    const float* ln_b     = (const float*)d_in[8];
    const float* ffn_w1   = (const float*)d_in[9];
    const float* ffn_b1   = (const float*)d_in[10];
    const float* ffn_w2   = (const float*)d_in[11];
    const float* ffn_b2   = (const float*)d_in[12];
    const float* ffn_ln_g = (const float*)d_in[13];
    const float* ffn_ln_b = (const float*)d_in[14];
    const float* cls_w1   = (const float*)d_in[15];
    const float* cls_b1   = (const float*)d_in[16];
    const float* cls_w2   = (const float*)d_in[17];
    const float* cls_b2   = (const float*)d_in[18];
    const int* src = ei;
    const int* dst = ei + EE;

    char* ws = (char*)d_ws;
    size_t off = 0;
    auto alloc = [&](size_t bytes) -> char* {
        char* p = ws + off;
        off = (off + bytes + 255) & ~(size_t)255;
        return p;
    };
    float* deg  = (float*)alloc((size_t)NN * 4);
    float* isq  = (float*)alloc((size_t)NN * 4);
    float* sn   = (float*)alloc((size_t)NN * 4);
    float* h    = (float*)alloc((size_t)NN * CC * 4);
    float* hw   = (float*)alloc((size_t)NN * CC * 4);
    float* agg  = (float*)alloc((size_t)NN * CC * 4);
    float* tt   = (float*)alloc((size_t)NN * CC * 4);
    float* g    = (float*)alloc((size_t)BB * CC * 4);
    float* clst = (float*)alloc((size_t)BB * CC * 4);
    float* cnt  = (float*)alloc((size_t)BB * 4);
    uint32_t* wp = (uint32_t*)alloc((size_t)11 * 8192 * 4);
    // wp layout: [0]=pre, [1..3]=conv, [4..6]=ffn1, [7..9]=ffn2, [10]=cls1

    const int T = 256;
    // Weight packing (bf16, WMMA B-fragment order)
    prep_weights<<<(1 * 8192 + T - 1) / T, T, 0, stream>>>(pre_w,  wp + 0 * 8192, 1);
    prep_weights<<<(3 * 8192 + T - 1) / T, T, 0, stream>>>(conv_w, wp + 1 * 8192, 3);
    prep_weights<<<(3 * 8192 + T - 1) / T, T, 0, stream>>>(ffn_w1, wp + 4 * 8192, 3);
    prep_weights<<<(3 * 8192 + T - 1) / T, T, 0, stream>>>(ffn_w2, wp + 7 * 8192, 3);
    prep_weights<<<(1 * 8192 + T - 1) / T, T, 0, stream>>>(cls_w1, wp + 10 * 8192, 1);

    // Degrees + normalization
    deg_init  <<<(NN + T - 1) / T, T, 0, stream>>>(deg, NN);
    deg_count <<<(EE + T - 1) / T, T, 0, stream>>>(dst, deg, EE);
    deg_finish<<<(NN + T - 1) / T, T, 0, stream>>>(deg, isq, sn, NN);

    const int GN = NN / 16;           // 2500 blocks per node-GEMM
    const int GE = (EE * 32) / T;     // 80000
    const int GC = (NN * 32) / T;     // 5000
    const int GL = NN / 8;            // 5000

    // h = x @ pre_w + pre_b
    gemm_bf16<true, false, false, false><<<GN, T, 0, stream>>>(
        x, wp, pre_b, nullptr, nullptr, nullptr, h, nullptr);

    for (int i = 0; i < LL; ++i) {
        // hw = h @ conv_w[i] ; agg = hw*selfnorm + conv_b (fused)
        gemm_bf16<false, false, false, true><<<GN, T, 0, stream>>>(
            h, wp + (1 + i) * 8192, nullptr, nullptr, sn, conv_b + i * CC, hw, agg);
        // agg += scatter of hw[src]*norm_e over edges
        edge_aggregate<<<GE, T, 0, stream>>>(src, dst, hw, isq, agg, EE);
        if (i < LL - 1) {
            // tt = relu(LN(agg)) + h
            layernorm128<true><<<GL, T, 0, stream>>>(
                agg, ln_g + i * CC, ln_b + i * CC, h, tt);
            // hw = relu(tt @ w1 + b1)
            gemm_bf16<true, true, false, false><<<GN, T, 0, stream>>>(
                tt, wp + (4 + i) * 8192, ffn_b1 + i * CC, nullptr, nullptr, nullptr, hw, nullptr);
            // agg = hw @ w2 + b2 + tt
            gemm_bf16<true, false, true, false><<<GN, T, 0, stream>>>(
                hw, wp + (7 + i) * 8192, ffn_b2 + i * CC, tt, nullptr, nullptr, agg, nullptr);
            // h = LN(agg)
            layernorm128<false><<<GL, T, 0, stream>>>(
                agg, ffn_ln_g + i * CC, ffn_ln_b + i * CC, nullptr, h);
        }
    }
    // final features live in `agg` (layer 2 has no norm/FFN)

    // Global mean pool
    zero_f<<<(BB * CC + T - 1) / T, T, 0, stream>>>(g, BB * CC);
    zero_f<<<1, T, 0, stream>>>(cnt, BB);
    pool_count<<<(NN + T - 1) / T, T, 0, stream>>>(batch, cnt, NN);
    pool_sum<<<GC, T, 0, stream>>>(agg, batch, g, NN);
    pool_div<<<(BB * CC + T - 1) / T, T, 0, stream>>>(g, cnt, BB * CC);

    // Classifier: clst = relu(g @ cls_w1 + cls_b1); out = clst @ cls_w2 + cls_b2
    gemm_bf16<true, true, false, false><<<BB / 16, T, 0, stream>>>(
        g, wp + 10 * 8192, cls_b1, nullptr, nullptr, nullptr, clst, nullptr);
    cls_out_kernel<<<BB / 8, T, 0, stream>>>(clst, cls_w2, cls_b2, (float*)d_out);
}
